// BestRqFramework_19980187861915
// MI455X (gfx1250) — matrix operand order repaired and checked by reading, not compiled, not run on hardware
//
#include <hip/hip_runtime.h>

typedef __attribute__((ext_vector_type(16))) __bf16 v16bf;
typedef __attribute__((ext_vector_type(8)))  float  v8f;

union Frag16 { uint4 u[2]; v16bf v; };
struct __align__(8) US4 { unsigned short x, y, z, w; };

#define N_TOK (8 * 8192)
#define DIM_D 512
#define DIM_Q 1024
#define DIM_C 4096
#define LDS_STRIDE 40   // 32 bf16 + 8 pad; 80B = 5*16B keeps 16B alignment

__device__ __forceinline__ unsigned short f2bf(float f) {
    unsigned int u = __float_as_uint(f);
    unsigned int r = u + 0x7fffu + ((u >> 16) & 1u);   // round-to-nearest-even
    return (unsigned short)(r >> 16);
}

// ---------------- CDNA5 async copy: global -> LDS, tracked by ASYNCcnt ----------------
__device__ __forceinline__ void async_ld_b128(unsigned lds_off, const void* gptr) {
    // GV mode: 64-bit global address in VGPR pair, LDS byte address in VDST VGPR.
    asm volatile("global_load_async_to_lds_b128 %0, %1, off"
                 :: "v"(lds_off), "v"(gptr) : "memory");
}

template <int N>
__device__ __forceinline__ void wait_async() {
    asm volatile("s_wait_asynccnt %c0" :: "i"(N) : "memory");
}

// Stage a ROWSx32 bf16 panel (row stride ld) into LDS [ROWS][LDS_STRIDE] via async DMA.
// ROWS*4 16B-chunks spread over 256 threads -> ROWS/64 async ops per thread.
template <int ROWS>
__device__ __forceinline__ void stage_panel_async(const unsigned short* __restrict__ src,
                                                  int ld, unsigned short* lds, int tid)
{
#pragma unroll
    for (int it = 0; it < ROWS / 64; ++it) {
        int c   = tid + it * 256;
        int row = c >> 2;
        int off = (c & 3) * 8;
        async_ld_b128((unsigned)(uintptr_t)(lds + row * LDS_STRIDE + off),
                      src + (size_t)row * ld + off);
    }
}

// ---------------------------------------------------------------- LayerNorm
__global__ __launch_bounds__(128) void ln_kernel(
    const float* __restrict__ x, const float* __restrict__ gamma,
    const float* __restrict__ beta, unsigned short* __restrict__ xb)
{
    const int row = blockIdx.x;
    const int tid = threadIdx.x;
    const int wave = tid >> 5, lane = tid & 31;
    const float* xr = x + (size_t)row * DIM_D;

    float4 v = ((const float4*)xr)[tid];          // 128 threads * 4 = 512
    float s  = v.x + v.y + v.z + v.w;
    float s2 = v.x * v.x + v.y * v.y + v.z * v.z + v.w * v.w;
    for (int off = 16; off > 0; off >>= 1) {
        s  += __shfl_xor(s,  off);
        s2 += __shfl_xor(s2, off);
    }
    __shared__ float ps[4], ps2[4];
    if (lane == 0) { ps[wave] = s; ps2[wave] = s2; }
    __syncthreads();
    s  = ps[0]  + ps[1]  + ps[2]  + ps[3];
    s2 = ps2[0] + ps2[1] + ps2[2] + ps2[3];

    const float mu   = s * (1.0f / DIM_D);
    const float var  = s2 * (1.0f / DIM_D) - mu * mu;
    const float rinv = rsqrtf(var + 1e-5f);

    float4 g  = ((const float4*)gamma)[tid];
    float4 be = ((const float4*)beta)[tid];
    US4 o;
    o.x = f2bf((v.x - mu) * rinv * g.x + be.x);
    o.y = f2bf((v.y - mu) * rinv * g.y + be.y);
    o.z = f2bf((v.z - mu) * rinv * g.z + be.z);
    o.w = f2bf((v.w - mu) * rinv * g.w + be.w);
    ((US4*)(xb + (size_t)row * DIM_D))[tid] = o;
}

// ---------------------------------------------------------------- converts
__global__ __launch_bounds__(256) void cvt_kernel(
    const float* __restrict__ src, unsigned short* __restrict__ dst, int n)
{
    int i = blockIdx.x * 256 + threadIdx.x;
    if (i < n) dst[i] = f2bf(src[i]);
}

// code_book (Q,C) f32 -> cbt (C,Q) bf16 via 32x32 LDS tile transpose
__global__ __launch_bounds__(256) void cbt_kernel(
    const float* __restrict__ cb, unsigned short* __restrict__ cbt)
{
    __shared__ unsigned short t[32][33];
    const int c0 = blockIdx.x * 32, q0 = blockIdx.y * 32;
    const int lx = threadIdx.x, ly = threadIdx.y;     // 32 x 8
    for (int i = 0; i < 32; i += 8)
        t[ly + i][lx] = f2bf(cb[(size_t)(q0 + ly + i) * DIM_C + c0 + lx]);
    __syncthreads();
    for (int i = 0; i < 32; i += 8)
        cbt[(size_t)(c0 + ly + i) * DIM_Q + q0 + lx] = t[lx][ly + i];
}

__global__ __launch_bounds__(256) void csq_kernel(
    const float* __restrict__ cb, float* __restrict__ csq)
{
    int c = blockIdx.x * 256 + threadIdx.x;
    float s = 0.f;
    for (int q = 0; q < DIM_Q; ++q) {
        float v = cb[(size_t)q * DIM_C + c];
        s += v * v;
    }
    csq[c] = s;
}

// ---------------------------------------------------------------- WMMA fragment loads
// A fragment 16x32 bf16: lane holds row M=lane&15; K chunks {h..h+7, h+16..h+23}, h=(lane<16?0:8)
__device__ __forceinline__ v16bf load_a_frag(const unsigned short* As, int rowBase, int lane)
{
    int m  = rowBase + (lane & 15);
    int kh = (lane < 16) ? 0 : 8;
    Frag16 f;
    f.u[0] = *(const uint4*)(As + m * LDS_STRIDE + kh);
    f.u[1] = *(const uint4*)(As + m * LDS_STRIDE + kh + 16);
    return f.v;
}

// B fragment 32x16 bf16: lane holds col N=lane&15; K chunks {h..h+7, h+16..h+23}, h=(lane<16?0:8)
__device__ __forceinline__ v16bf load_b_frag(const unsigned short* Bs, int colBase, int lane)
{
    int n  = colBase + (lane & 15);
    int k0 = (lane < 16) ? 0 : 8;
    Frag16 f;
    f.u[0] = *(const uint4*)(Bs + n * LDS_STRIDE + k0);
    f.u[1] = *(const uint4*)(Bs + n * LDS_STRIDE + k0 + 16);
    return f.v;
}

__device__ __forceinline__ v8f zero8()
{
    v8f z;
#pragma unroll
    for (int i = 0; i < 8; ++i) z[i] = 0.f;
    return z;
}

// ---------------------------------------------------------------- GEMM: out = X * W^T (+bias)
// Block tile 256x128, 8 waves in 4x2 grid, each wave 4x4 16x16 tiles (128 acc VGPRs).
// Double-buffered async-to-LDS K-panel pipeline (6 async ops/wave/stage).
template <int OUT_BF16>
__global__ __launch_bounds__(256) void gemm_kernel(
    const unsigned short* __restrict__ X, const unsigned short* __restrict__ W,
    const float* __restrict__ bias, void* __restrict__ out, int K, int ldo)
{
    __shared__ __align__(16) unsigned short As[2][256 * LDS_STRIDE];
    __shared__ __align__(16) unsigned short Bs[2][128 * LDS_STRIDE];

    const int tid  = threadIdx.x;
    const int wave = tid >> 5, lane = tid & 31;
    const int wm = wave >> 1, wn = wave & 1;          // 4 x 2 wave grid
    const int rowBlk = blockIdx.x * 256;
    const int colBlk = blockIdx.y * 128;

    const unsigned short* Xp = X + (size_t)rowBlk * K;
    const unsigned short* Wp = W + (size_t)colBlk * K;

    v8f acc[4][4];
#pragma unroll
    for (int mi = 0; mi < 4; ++mi)
#pragma unroll
        for (int ni = 0; ni < 4; ++ni) acc[mi][ni] = zero8();

    stage_panel_async<256>(Xp, K, As[0], tid);
    stage_panel_async<128>(Wp, K, Bs[0], tid);

    const int nk = K >> 5;
    for (int ks = 0; ks < nk; ++ks) {
        const int cur = ks & 1;
        __syncthreads();                               // readers of buf cur^1 are done
        if (ks + 1 < nk) {
            stage_panel_async<256>(Xp + (ks + 1) * 32, K, As[cur ^ 1], tid);
            stage_panel_async<128>(Wp + (ks + 1) * 32, K, Bs[cur ^ 1], tid);
            wait_async<6>();                           // stage ks landed (in-order)
        } else {
            wait_async<0>();
        }
        __syncthreads();                               // all waves' stage ks data present

        v16bf a[4], b[4];
#pragma unroll
        for (int mi = 0; mi < 4; ++mi) a[mi] = load_a_frag(As[cur], wm * 64 + mi * 16, lane);
#pragma unroll
        for (int ni = 0; ni < 4; ++ni) b[ni] = load_b_frag(Bs[cur], wn * 64 + ni * 16, lane);
#pragma unroll
        for (int mi = 0; mi < 4; ++mi)
#pragma unroll
            for (int ni = 0; ni < 4; ++ni)
                acc[mi][ni] = __builtin_amdgcn_wmma_f32_16x16x32_bf16(
                    false, a[mi], false, b[ni], (short)0, acc[mi][ni], false, false);
    }

    const int hi = (lane >> 4) << 3;   // M offset 0 or 8
#pragma unroll
    for (int mi = 0; mi < 4; ++mi) {
#pragma unroll
        for (int ni = 0; ni < 4; ++ni) {
            int n  = colBlk + wn * 64 + ni * 16 + (lane & 15);
            float bv = bias ? bias[n] : 0.f;
            int m0 = rowBlk + wm * 64 + mi * 16 + hi;
#pragma unroll
            for (int v = 0; v < 8; ++v) {
                float val = acc[mi][ni][v] + bv;
                if (OUT_BF16)
                    ((unsigned short*)out)[(size_t)(m0 + v) * ldo + n] = f2bf(val);
                else
                    ((float*)out)[(size_t)(m0 + v) * ldo + n] = val;
            }
        }
    }
}

// ---------------------------------------------------------------- fused cross-GEMM + argmin
// cross = targets(N,Q) * cbt(C,Q)^T ; labels = argmin_c (c_sq[c] - 2*cross)
// 128x128 tile per codebook block, async double-buffered K pipeline (4 ops/wave/stage).
__global__ __launch_bounds__(256) void argmin_kernel(
    const unsigned short* __restrict__ Tg, const unsigned short* __restrict__ CbT,
    const float* __restrict__ csq, float* __restrict__ labels)
{
    __shared__ __align__(16) unsigned short As[2][128 * LDS_STRIDE];
    __shared__ __align__(16) unsigned short Bs[2][128 * LDS_STRIDE];
    __shared__ float Ss[128 * 128];

    const int tid  = threadIdx.x;
    const int wave = tid >> 5, lane = tid & 31;
    const int wm = wave >> 1, wn = wave & 1;
    const int rowBlk = blockIdx.x * 128;
    const int hi = (lane >> 4) << 3;

    float best = 3.402823e38f;
    int bestIdx = 0;
    const int r  = tid & 127;           // row this thread scans
    const int ch = (tid >> 7) * 64;     // column half

    const unsigned short* Tp = Tg + (size_t)rowBlk * DIM_Q;

    for (int cb = 0; cb < DIM_C; cb += 128) {
        const unsigned short* Cp = CbT + (size_t)cb * DIM_Q;

        v8f acc[2][4];
#pragma unroll
        for (int mi = 0; mi < 2; ++mi)
#pragma unroll
            for (int ni = 0; ni < 4; ++ni) acc[mi][ni] = zero8();

        stage_panel_async<128>(Tp, DIM_Q, As[0], tid);
        stage_panel_async<128>(Cp, DIM_Q, Bs[0], tid);

        const int nk = DIM_Q >> 5;
        for (int ks = 0; ks < nk; ++ks) {
            const int cur = ks & 1;
            __syncthreads();
            if (ks + 1 < nk) {
                stage_panel_async<128>(Tp + (ks + 1) * 32, DIM_Q, As[cur ^ 1], tid);
                stage_panel_async<128>(Cp + (ks + 1) * 32, DIM_Q, Bs[cur ^ 1], tid);
                wait_async<4>();
            } else {
                wait_async<0>();
            }
            __syncthreads();

            v16bf a[2], b[4];
#pragma unroll
            for (int mi = 0; mi < 2; ++mi) a[mi] = load_a_frag(As[cur], wm * 32 + mi * 16, lane);
#pragma unroll
            for (int ni = 0; ni < 4; ++ni) b[ni] = load_b_frag(Bs[cur], wn * 64 + ni * 16, lane);
#pragma unroll
            for (int mi = 0; mi < 2; ++mi)
#pragma unroll
                for (int ni = 0; ni < 4; ++ni)
                    acc[mi][ni] = __builtin_amdgcn_wmma_f32_16x16x32_bf16(
                        false, a[mi], false, b[ni], (short)0, acc[mi][ni], false, false);
        }

        // scores -> LDS, then per-row running argmin
#pragma unroll
        for (int mi = 0; mi < 2; ++mi)
#pragma unroll
            for (int ni = 0; ni < 4; ++ni) {
                int col  = wn * 64 + ni * 16 + (lane & 15);
                int row0 = wm * 32 + mi * 16 + hi;
#pragma unroll
                for (int v = 0; v < 8; ++v)
                    Ss[(row0 + v) * 128 + col] = acc[mi][ni][v];
            }
        __syncthreads();

        for (int j = 0; j < 64; ++j) {
            int c = cb + ch + j;
            float s = csq[c] - 2.0f * Ss[r * 128 + ch + j];
            if (s < best) { best = s; bestIdx = c; }
        }
        __syncthreads();   // scans done before Ss / LDS bufs are overwritten
    }

    // combine the two column-half partials per row through LDS
    if (tid >= 128) {
        Ss[tid - 128]       = best;
        Ss[128 + tid - 128] = __int_as_float(bestIdx);
    }
    __syncthreads();
    if (tid < 128) {
        float ov = Ss[tid];
        int   oi = __float_as_int(Ss[128 + tid]);
        if (ov < best) { best = ov; bestIdx = oi; }
        labels[rowBlk + tid] = (float)bestIdx;
    }
}

// ---------------------------------------------------------------- launch
extern "C" void kernel_launch(void* const* d_in, const int* in_sizes, int n_in,
                              void* d_out, int out_size, void* d_ws, size_t ws_size,
                              hipStream_t stream)
{
    const float* input    = (const float*)d_in[0];
    const float* gamma    = (const float*)d_in[1];
    const float* beta     = (const float*)d_in[2];
    const float* projw    = (const float*)d_in[3];
    const float* codebook = (const float*)d_in[4];
    const float* encw     = (const float*)d_in[5];
    const float* encb     = (const float*)d_in[6];

    char* ws = (char*)d_ws;
    unsigned short* xb  = (unsigned short*)ws;                         // N*D bf16 (64 MB)
    unsigned short* pwb = (unsigned short*)(ws + (size_t)N_TOK * DIM_D * 2);
    unsigned short* ewb = pwb + (size_t)DIM_Q * DIM_D;
    unsigned short* cbt = ewb + (size_t)DIM_Q * DIM_D;                 // (C,Q) bf16 (8 MB)
    float*          csq = (float*)(cbt + (size_t)DIM_C * DIM_Q);       // C f32
    unsigned short* tgt = (unsigned short*)((char*)csq + (size_t)DIM_C * 4); // N*Q bf16 (128 MB)

    float* enc_out = (float*)d_out;
    float* labels  = enc_out + (size_t)N_TOK * DIM_Q;

    ln_kernel<<<N_TOK, 128, 0, stream>>>(input, gamma, beta, xb);
    cvt_kernel<<<(DIM_Q * DIM_D + 255) / 256, 256, 0, stream>>>(projw, pwb, DIM_Q * DIM_D);
    cvt_kernel<<<(DIM_Q * DIM_D + 255) / 256, 256, 0, stream>>>(encw, ewb, DIM_Q * DIM_D);
    cbt_kernel<<<dim3(DIM_C / 32, DIM_Q / 32), dim3(32, 8), 0, stream>>>(codebook, cbt);
    csq_kernel<<<DIM_C / 256, 256, 0, stream>>>(codebook, csq);

    dim3 ggrid(N_TOK / 256, DIM_Q / 128);
    gemm_kernel<0><<<ggrid, 256, 0, stream>>>(xb, ewb, encb, enc_out, DIM_D, DIM_Q);
    gemm_kernel<1><<<ggrid, 256, 0, stream>>>(xb, pwb, nullptr, tgt, DIM_D, DIM_Q);

    argmin_kernel<<<N_TOK / 128, 256, 0, stream>>>(tgt, cbt, csq, labels);
}
